// RealGatedLRU_88854283420038
// MI455X (gfx1250) — compile-verified, matrix operand order checked by reading, not run on hardware
//
#include <hip/hip_runtime.h>

// ---------------------------------------------------------------------------
// RealGatedLRU on MI455X (gfx1250, wave32, WMMA, async-to-LDS staging).
//
// Pipeline:
//   1. f32->bf16 convert of inputs + 4 weight matrices (one-time, cheap)
//   2. csp[h] = C * softplus(pre_a[h])
//   3. Fused gate GEMM: 3 GEMMs (Wr,Wi,Wx) sharing one LDS-staged A-tile.
//      Tiles staged with global_load_async_to_lds_b128 (double-buffered,
//      ASYNCcnt-tracked), fragments read with ds_load_b128, math via
//      v_wmma_f32_16x16x32_bf16. Full elementwise gate math in the epilogue.
//   4. Chunked linear scan (3 passes): per-(channel,chunk) aggregate,
//      per-channel carry scan, final scan writing h as bf16.
//   5. Output GEMM: h_bf16 @ Wo^T + bo via bf16 WMMA -> f32 out.
// ---------------------------------------------------------------------------

#define kT     8192
#define kDIN   1024
#define kDH    2048
#define kDOUT  1024
#define kC     8.0f

#define NCHUNK 64
#define TCHUNK 128   // NCHUNK * TCHUNK == kT

// LDS tile row stride in ushorts: 32 k-elements + 8 pad = 40 (80B; 20 banks,
// coprime walk across 16 lanes -> conflict-free ds_load_b128 fragment reads)
#define LDS_STRIDE 40

typedef __attribute__((ext_vector_type(16))) __bf16 v16bf;
typedef __attribute__((ext_vector_type(8)))  float  v8f;
typedef __attribute__((ext_vector_type(4)))  int    v4i;

union FragU {
    struct { v4i lo, hi; } i;
    v16bf v;
};

__device__ __forceinline__ unsigned short f32_to_bf16_rne(float f) {
    unsigned u = __float_as_uint(f);
    u += 0x7FFFu + ((u >> 16) & 1u);
    return (unsigned short)(u >> 16);
}

// Async 16B copy global -> LDS (per-lane), tracked by ASYNCcnt.
__device__ __forceinline__ void async_copy_b128(unsigned lds_off,
                                                const unsigned short* gptr) {
    asm volatile("global_load_async_to_lds_b128 %0, %1, off"
                 :: "v"(lds_off), "v"(gptr) : "memory");
}

__device__ __forceinline__ void wait_async_zero() {
    asm volatile("s_wait_asynccnt 0x0" ::: "memory");
}

// Fragment load (16 bf16 per lane) from an LDS tile row pointer.
// Per CDNA5 ISA 16-bit matrix layout:
//   lanes 0-15 : row/col (base + lane),     K = kk+{0..7}  and kk+{16..23}
//   lanes16-31 : row/col (base + lane-16),  K = kk+{8..15} and kk+{24..31}
__device__ __forceinline__ v16bf load_frag_lds(const unsigned short* p) {
    FragU u;
    u.i.lo = *reinterpret_cast<const v4i*>(p);        // K = kh .. kh+7
    u.i.hi = *reinterpret_cast<const v4i*>(p + 16);   // K = kh+16 .. kh+23
    return u.v;
}

// Direct-from-global fragment load (used by the output GEMM).
__device__ __forceinline__ v16bf load_frag_bf16(const unsigned short* __restrict__ base,
                                                int row0, int ld, int kk, int lane) {
    const int l  = lane & 15;
    const int kh = (lane >> 4) << 3;  // 0 or 8
    const unsigned short* p = base + (size_t)(row0 + l) * (size_t)ld + (size_t)(kk + kh);
    FragU u;
    u.i.lo = *reinterpret_cast<const v4i*>(p);
    u.i.hi = *reinterpret_cast<const v4i*>(p + 16);
    return u.v;
}

// ---------------------------------------------------------------------------
// 1) f32 -> bf16 conversion (grid-stride)
// ---------------------------------------------------------------------------
__global__ void cvt_bf16_kernel(const float* __restrict__ in,
                                unsigned short* __restrict__ out, int n) {
    int i = blockIdx.x * blockDim.x + threadIdx.x;
    int stride = gridDim.x * blockDim.x;
    for (; i < n; i += stride) out[i] = f32_to_bf16_rne(in[i]);
}

// ---------------------------------------------------------------------------
// 2) csp[h] = C * softplus(pre_a[h])
// ---------------------------------------------------------------------------
__global__ void prep_csp_kernel(const float* __restrict__ pre_a,
                                float* __restrict__ csp) {
    int h = blockIdx.x * blockDim.x + threadIdx.x;
    if (h < kDH) csp[h] = kC * log1pf(__expf(pre_a[h]));
}

// ---------------------------------------------------------------------------
// 3) Fused triple gate GEMM + elementwise epilogue.
//    Block tile: 128 (T) x 64 (D_HID); 8 waves as 4x2; wave tile 32x32.
//    Per wave: 12 accumulators (2x2 tiles x 3 weight matrices).
//    A-tile + 3 weight tiles double-buffered in LDS via async copies.
// ---------------------------------------------------------------------------
__global__ __launch_bounds__(256)
void gates_wmma_kernel(const unsigned short* __restrict__ Xbf,
                       const unsigned short* __restrict__ Wrb,
                       const unsigned short* __restrict__ Wib,
                       const unsigned short* __restrict__ Wxb,
                       const float* __restrict__ br,
                       const float* __restrict__ bi,
                       const float* __restrict__ bx,
                       const float* __restrict__ csp,
                       float* __restrict__ a_out,
                       float* __restrict__ g_out) {
    __shared__ __align__(16) unsigned short tA[2][128 * LDS_STRIDE];      // 20 KB
    __shared__ __align__(16) unsigned short tB[3][2][64 * LDS_STRIDE];    // 30 KB

    const int tid  = threadIdx.x;
    const int lane = tid & 31;
    const int wave = tid >> 5;
    const int wm = wave >> 1;             // 0..3
    const int wn = wave & 1;              // 0..1
    const int mblk = blockIdx.x * 128;
    const int nblk = blockIdx.y * 64;

    const unsigned short* gw[3] = { Wrb, Wib, Wxb };

    // Cooperative async staging of one k-slice (32 deep) into buffer `buf`.
    auto stage = [&](int kk, int buf) {
        // A tile: 128 rows x 32 k = 512 x 16B chunks; 2 per thread
        #pragma unroll
        for (int r = 0; r < 2; ++r) {
            const int c   = tid + 256 * r;
            const int row = c >> 2;
            const int q   = c & 3;
            const unsigned short* g =
                Xbf + (size_t)(mblk + row) * kDIN + (size_t)(kk + q * 8);
            const unsigned lds =
                (unsigned)(size_t)&tA[buf][row * LDS_STRIDE + q * 8];
            async_copy_b128(lds, g);
        }
        // 3 weight tiles: 64 rows x 32 k = 256 x 16B chunks; 1 per thread each
        const int row = tid >> 2;
        const int q   = tid & 3;
        #pragma unroll
        for (int w = 0; w < 3; ++w) {
            const unsigned short* g =
                gw[w] + (size_t)(nblk + row) * kDIN + (size_t)(kk + q * 8);
            const unsigned lds =
                (unsigned)(size_t)&tB[w][buf][row * LDS_STRIDE + q * 8];
            async_copy_b128(lds, g);
        }
    };

    v8f aR[2][2] = {}; v8f aI[2][2] = {}; v8f aX[2][2] = {};

    const int l  = lane & 15;
    const int kh = (lane >> 4) << 3;  // 0 or 8

    int buf = 0;
    stage(0, 0);
    for (int kk = 0; kk < kDIN; kk += 32) {
        wait_async_zero();      // my staged chunks for `buf` have landed
        __syncthreads();        // everyone's chunks have landed; prev reads done
        if (kk + 32 < kDIN) stage(kk + 32, buf ^ 1);

        v16bf af[2];
        #pragma unroll
        for (int i = 0; i < 2; ++i)
            af[i] = load_frag_lds(&tA[buf][(wm * 32 + 16 * i + l) * LDS_STRIDE + kh]);

        v16bf fr[2], fi[2], fx[2];
        #pragma unroll
        for (int j = 0; j < 2; ++j) {
            const int rr = (wn * 32 + 16 * j + l) * LDS_STRIDE + kh;
            fr[j] = load_frag_lds(&tB[0][buf][rr]);
            fi[j] = load_frag_lds(&tB[1][buf][rr]);
            fx[j] = load_frag_lds(&tB[2][buf][rr]);
        }

        #pragma unroll
        for (int i = 0; i < 2; ++i) {
            #pragma unroll
            for (int j = 0; j < 2; ++j) {
                aR[i][j] = __builtin_amdgcn_wmma_f32_16x16x32_bf16(
                    false, af[i], false, fr[j], (short)0, aR[i][j], false, false);
                aI[i][j] = __builtin_amdgcn_wmma_f32_16x16x32_bf16(
                    false, af[i], false, fi[j], (short)0, aI[i][j], false, false);
                aX[i][j] = __builtin_amdgcn_wmma_f32_16x16x32_bf16(
                    false, af[i], false, fx[j], (short)0, aX[i][j], false, false);
            }
        }
        buf ^= 1;
    }

    // Epilogue: bias + sigmoid gates + decay + gated input, write a_t and gx.
    // C/D layout: elem r of v8f -> m = m_tile + r + 8*(lane>=16), n = n_tile + (lane&15)
    const int m_base = mblk + wm * 32;
    const int n_base = nblk + wn * 32;
    #pragma unroll
    for (int j = 0; j < 2; ++j) {
        const int n  = n_base + 16 * j + l;
        const float bR = br[n], bI = bi[n], bX = bx[n], cs = csp[n];
        #pragma unroll
        for (int i = 0; i < 2; ++i) {
            const int m0 = m_base + 16 * i + kh;
            #pragma unroll
            for (int r = 0; r < 8; ++r) {
                const int m = m0 + r;
                const float pr = aR[i][j][r] + bR;
                const float pi = aI[i][j][r] + bI;
                const float px = aX[i][j][r] + bX;
                const float rg = 1.0f / (1.0f + __expf(-pr));
                const float ig = 1.0f / (1.0f + __expf(-pi));
                const float a  = __expf(-cs * rg);
                const float om = fmaxf(fmaf(-a, a, 1.0f), 0.0f);
                const float gx = sqrtf(om) * ig * px;
                const size_t idx = (size_t)m * kDH + (size_t)n;
                a_out[idx] = a;
                g_out[idx] = gx;
            }
        }
    }
}

// ---------------------------------------------------------------------------
// 4) Chunked linear scan: h_t = a_t * h_{t-1} + gx_t
// ---------------------------------------------------------------------------
__global__ void scan_pass1_kernel(const float* __restrict__ a_buf,
                                  const float* __restrict__ g_buf,
                                  float* __restrict__ chunkA,
                                  float* __restrict__ chunkB) {
    const int idx = blockIdx.x * blockDim.x + threadIdx.x;
    const int ch = idx % kDH;
    const int ck = idx / kDH;
    if (ck >= NCHUNK) return;
    const int t0 = ck * TCHUNK;
    float A = 1.0f, B = 0.0f;
    for (int t = 0; t < TCHUNK; ++t) {
        const size_t p = (size_t)(t0 + t) * kDH + (size_t)ch;
        const float a = a_buf[p];
        B = fmaf(a, B, g_buf[p]);
        A *= a;
    }
    chunkA[(size_t)ck * kDH + ch] = A;
    chunkB[(size_t)ck * kDH + ch] = B;
}

__global__ void scan_pass2_kernel(const float* __restrict__ chunkA,
                                  const float* __restrict__ chunkB,
                                  float* __restrict__ carry) {
    const int ch = blockIdx.x * blockDim.x + threadIdx.x;
    if (ch >= kDH) return;
    float s = 0.0f;
    for (int ck = 0; ck < NCHUNK; ++ck) {
        const size_t p = (size_t)ck * kDH + ch;
        carry[p] = s;
        s = fmaf(chunkA[p], s, chunkB[p]);
    }
}

__global__ void scan_pass3_kernel(const float* __restrict__ a_buf,
                                  const float* __restrict__ g_buf,
                                  const float* __restrict__ carry,
                                  unsigned short* __restrict__ h_bf) {
    const int idx = blockIdx.x * blockDim.x + threadIdx.x;
    const int ch = idx % kDH;
    const int ck = idx / kDH;
    if (ck >= NCHUNK) return;
    const int t0 = ck * TCHUNK;
    float h = carry[(size_t)ck * kDH + ch];
    for (int t = 0; t < TCHUNK; ++t) {
        const size_t p = (size_t)(t0 + t) * kDH + (size_t)ch;
        h = fmaf(a_buf[p], h, g_buf[p]);
        h_bf[p] = f32_to_bf16_rne(h);
    }
}

// ---------------------------------------------------------------------------
// 5) Output GEMM: out = H(bf16) @ Wo^T + bo.
//    Block tile 128 x 128; 8 waves as 4x2; wave tile 32x64 (2x4 WMMA tiles).
// ---------------------------------------------------------------------------
__global__ __launch_bounds__(256)
void out_wmma_kernel(const unsigned short* __restrict__ Hbf,
                     const unsigned short* __restrict__ Wob,
                     const float* __restrict__ bo,
                     float* __restrict__ out) {
    const int lane = threadIdx.x & 31;
    const int wave = threadIdx.x >> 5;
    const int wm = wave >> 1;             // 0..3
    const int wn = wave & 1;              // 0..1
    const int m_base = blockIdx.x * 128 + wm * 32;
    const int n_base = blockIdx.y * 128 + wn * 64;

    v8f acc[2][4] = {};

    for (int kk = 0; kk < kDH; kk += 32) {
        v16bf af[2];
        #pragma unroll
        for (int i = 0; i < 2; ++i)
            af[i] = load_frag_bf16(Hbf, m_base + 16 * i, kDH, kk, lane);
        v16bf bf_[4];
        #pragma unroll
        for (int j = 0; j < 4; ++j)
            bf_[j] = load_frag_bf16(Wob, n_base + 16 * j, kDH, kk, lane);

        #pragma unroll
        for (int i = 0; i < 2; ++i) {
            #pragma unroll
            for (int j = 0; j < 4; ++j) {
                acc[i][j] = __builtin_amdgcn_wmma_f32_16x16x32_bf16(
                    false, af[i], false, bf_[j], (short)0, acc[i][j], false, false);
            }
        }
    }

    #pragma unroll
    for (int j = 0; j < 4; ++j) {
        const int n  = n_base + 16 * j + (lane & 15);
        const float b = bo[n];
        #pragma unroll
        for (int i = 0; i < 2; ++i) {
            const int m0 = m_base + 16 * i + ((lane >> 4) << 3);
            #pragma unroll
            for (int r = 0; r < 8; ++r) {
                out[(size_t)(m0 + r) * kDOUT + (size_t)n] = acc[i][j][r] + b;
            }
        }
    }
}

// ---------------------------------------------------------------------------
// Launch
// ---------------------------------------------------------------------------
extern "C" void kernel_launch(void* const* d_in, const int* in_sizes, int n_in,
                              void* d_out, int out_size, void* d_ws, size_t ws_size,
                              hipStream_t stream) {
    const float* x     = (const float*)d_in[0];   // [T, D_IN]
    const float* pre_a = (const float*)d_in[1];   // [D_HID]
    const float* Wr    = (const float*)d_in[2];   // [D_HID, D_IN]
    const float* br    = (const float*)d_in[3];
    const float* Wi    = (const float*)d_in[4];
    const float* bi    = (const float*)d_in[5];
    const float* Wx    = (const float*)d_in[6];
    const float* bx    = (const float*)d_in[7];
    const float* Wo    = (const float*)d_in[8];   // [D_OUT, D_HID]
    const float* bo    = (const float*)d_in[9];
    float* out = (float*)d_out;

    // Workspace carving (all offsets 256B aligned)
    char* ws = (char*)d_ws;
    size_t off = 0;
    auto carve = [&](size_t bytes) {
        void* p = ws + off;
        off += (bytes + 255) & ~(size_t)255;
        return p;
    };
    unsigned short* Xbf  = (unsigned short*)carve((size_t)kT * kDIN * 2);
    unsigned short* Wrb  = (unsigned short*)carve((size_t)kDH * kDIN * 2);
    unsigned short* Wib  = (unsigned short*)carve((size_t)kDH * kDIN * 2);
    unsigned short* Wxb  = (unsigned short*)carve((size_t)kDH * kDIN * 2);
    unsigned short* Wob  = (unsigned short*)carve((size_t)kDOUT * kDH * 2);
    float*          csp  = (float*)carve((size_t)kDH * 4);
    float*          aBuf = (float*)carve((size_t)kT * kDH * 4);
    float*          gBuf = (float*)carve((size_t)kT * kDH * 4);
    unsigned short* Hbf  = (unsigned short*)carve((size_t)kT * kDH * 2);
    float*          cA   = (float*)carve((size_t)NCHUNK * kDH * 4);
    float*          cB   = (float*)carve((size_t)NCHUNK * kDH * 4);
    float*          cry  = (float*)carve((size_t)NCHUNK * kDH * 4);
    (void)ws_size; (void)n_in; (void)in_sizes; (void)out_size;

    // 1) bf16 conversions
    cvt_bf16_kernel<<<2048, 256, 0, stream>>>(x,  Xbf, kT * kDIN);
    cvt_bf16_kernel<<<1024, 256, 0, stream>>>(Wr, Wrb, kDH * kDIN);
    cvt_bf16_kernel<<<1024, 256, 0, stream>>>(Wi, Wib, kDH * kDIN);
    cvt_bf16_kernel<<<1024, 256, 0, stream>>>(Wx, Wxb, kDH * kDIN);
    cvt_bf16_kernel<<<1024, 256, 0, stream>>>(Wo, Wob, kDOUT * kDH);

    // 2) decay prefactor
    prep_csp_kernel<<<kDH / 256, 256, 0, stream>>>(pre_a, csp);

    // 3) fused gate GEMMs + elementwise (async-LDS staged)
    gates_wmma_kernel<<<dim3(kT / 128, kDH / 64), 256, 0, stream>>>(
        Xbf, Wrb, Wib, Wxb, br, bi, bx, csp, aBuf, gBuf);

    // 4) chunked scan
    scan_pass1_kernel<<<(kDH * NCHUNK) / 256, 256, 0, stream>>>(aBuf, gBuf, cA, cB);
    scan_pass2_kernel<<<kDH / 256, 256, 0, stream>>>(cA, cB, cry);
    scan_pass3_kernel<<<(kDH * NCHUNK) / 256, 256, 0, stream>>>(aBuf, gBuf, cry, Hbf);

    // 5) output GEMM
    out_wmma_kernel<<<dim3(kT / 128, kDOUT / 128), 256, 0, stream>>>(Hbf, Wob, bo, out);
}
